// BahdanauMonotonicAttention_39831526703180
// MI455X (gfx1250) — matrix operand order repaired and checked by reading, not compiled, not run on hardware
//
#include <hip/hip_runtime.h>
#include <hip/hip_bf16.h>

// Problem dims (fixed by the reference)
#define B_  32
#define T_  8192
#define QD  512
#define VD  256
#define DD  128

#define TINY_F 1.17549435e-38f

typedef __attribute__((ext_vector_type(16))) __bf16 v16bf;
typedef __attribute__((ext_vector_type(4)))  __bf16 v4bf;
typedef __attribute__((ext_vector_type(8)))  float  v8f;

#if __has_builtin(__builtin_amdgcn_tanhf)
#define FAST_TANH(x) __builtin_amdgcn_tanhf(x)
#else
#define FAST_TANH(x) tanhf(x)
#endif

#if __has_builtin(__builtin_amdgcn_rcpf)
#define FAST_RCP(x) __builtin_amdgcn_rcpf(x)
#else
#define FAST_RCP(x) (1.0f / (x))
#endif

// ---------------------------------------------------------------------------
// K0: q = query @ Wq + bq   [B, D]
// ---------------------------------------------------------------------------
__global__ __launch_bounds__(128) void k_q(const float* __restrict__ query,
                                           const float* __restrict__ Wq,
                                           const float* __restrict__ bq,
                                           float* __restrict__ q_out) {
  const int b = blockIdx.x;
  const int d = threadIdx.x;
  float s = bq[d];
  const float* qr = query + b * QD;
  #pragma unroll 4
  for (int k = 0; k < QD; ++k) s = fmaf(qr[k], Wq[k * DD + d], s);
  q_out[b * DD + d] = s;
}

// ---------------------------------------------------------------------------
// A-fragment: 16-bit A 16x32 layout — lane half h holds K in [h*8, h*8+8)
// and [16+h*8, 16+h*8+8); convert fp32 -> bf16 on the fly.
// ---------------------------------------------------------------------------
__device__ __forceinline__ v16bf load_a_frag(const float* __restrict__ r0) {
  const float4 a0 = *(const float4*)(r0);
  const float4 a1 = *(const float4*)(r0 + 4);
  const float4 a2 = *(const float4*)(r0 + 16);
  const float4 a3 = *(const float4*)(r0 + 20);
  v16bf f;
  f[0]  = (__bf16)a0.x; f[1]  = (__bf16)a0.y; f[2]  = (__bf16)a0.z; f[3]  = (__bf16)a0.w;
  f[4]  = (__bf16)a1.x; f[5]  = (__bf16)a1.y; f[6]  = (__bf16)a1.z; f[7]  = (__bf16)a1.w;
  f[8]  = (__bf16)a2.x; f[9]  = (__bf16)a2.y; f[10] = (__bf16)a2.z; f[11] = (__bf16)a2.w;
  f[12] = (__bf16)a3.x; f[13] = (__bf16)a3.y; f[14] = (__bf16)a3.z; f[15] = (__bf16)a3.w;
  return f;
}

// ---------------------------------------------------------------------------
// K1: v = value @ Wv + bv (bf16 WMMA), fused Bahdanau score.
// grid (T/256, B), 256 threads = 8 waves; each wave owns TWO 16-row M-tiles
// (32 t-rows) x full N=128, so every B-fragment LDS read feeds 2 WMMAs.
// ---------------------------------------------------------------------------
__global__ __launch_bounds__(256) void k_v_score(
    const float* __restrict__ value, const float* __restrict__ Wv,
    const float* __restrict__ bv, const float* __restrict__ av,
    const float* __restrict__ qvec, const float* __restrict__ sbias,
    __bf16* __restrict__ v_out, float* __restrict__ score_out) {
  // Wv staged transposed in LDS: WvT[d][k] (bf16, 64 KB)
  __shared__ __bf16 WvT[DD][VD];
  const int tid = threadIdx.x;
  for (int idx = tid; idx < DD * VD; idx += 256) {
    const int d = idx >> 8;        // 0..127
    const int k = idx & (VD - 1);  // 0..255
    WvT[d][k] = (__bf16)Wv[k * DD + d];
  }
  __syncthreads();

  const int b    = blockIdx.y;
  const int wave = tid >> 5;
  const int lane = tid & 31;
  const int half = lane >> 4;  // 0: lanes 0-15, 1: lanes 16-31
  const int l16  = lane & 15;
  const int t0   = blockIdx.x * 256 + wave * 32;

  v8f acc0[8], acc1[8];
  #pragma unroll
  for (int n = 0; n < 8; ++n) { acc0[n] = (v8f)0.0f; acc1[n] = (v8f)0.0f; }

  const float* vrow0 = value + ((size_t)b * T_ + (t0 + l16)) * VD;
  const float* vrow1 = vrow0 + (size_t)16 * VD;

  #pragma unroll 1
  for (int ks = 0; ks < 8; ++ks) {
    const int kbase = ks * 32;
    const v16bf afr0 = load_a_frag(vrow0 + kbase + half * 8);
    const v16bf afr1 = load_a_frag(vrow1 + kbase + half * 8);
    #pragma unroll
    for (int n = 0; n < 8; ++n) {
      // 16-bit B 32x16 layout: N = l16; lane half h holds K [h*8, h*8+8) and [16+h*8, 16+h*8+8)
      typedef __attribute__((ext_vector_type(8))) __bf16 v8bf_;
      const v8bf_ blo = *(const v8bf_*)&WvT[n * 16 + l16][kbase + half * 8];
      const v8bf_ bhi = *(const v8bf_*)&WvT[n * 16 + l16][kbase + 16 + half * 8];
      const v16bf bfr = __builtin_shufflevector(blo, bhi, 0, 1, 2, 3, 4, 5, 6, 7, 8, 9, 10, 11, 12, 13, 14, 15);
      acc0[n] = __builtin_amdgcn_wmma_f32_16x16x32_bf16(
          false, afr0, false, bfr, (short)0, acc0[n], false, false);
      acc1[n] = __builtin_amdgcn_wmma_f32_16x16x32_bf16(
          false, afr1, false, bfr, (short)0, acc1[n], false, false);
    }
  }

  // Epilogue constants: d = n*16 + l16 per n-tile
  float qv[8], avv[8], bvv[8];
  #pragma unroll
  for (int n = 0; n < 8; ++n) {
    const int d = n * 16 + l16;
    qv[n]  = qvec[b * DD + d];
    avv[n] = av[d];
    bvv[n] = bv[d];
  }
  const float sb = *sbias;

  // C layout: VGPR i -> row M = i + 8*half, col N = l16 (per n-tile).
  #pragma unroll
  for (int tile = 0; tile < 2; ++tile) {
    const v8f* acc = tile ? acc1 : acc0;
    const int tb = t0 + tile * 16;
    float sc[8];
    #pragma unroll
    for (int i = 0; i < 8; ++i) sc[i] = 0.0f;

    #pragma unroll
    for (int n = 0; n < 8; ++n) {
      const int d = n * 16 + l16;
      #pragma unroll
      for (int i = 0; i < 8; ++i) {
        const float vv = acc[n][i] + bvv[n];
        const int tr = tb + i + half * 8;
        v_out[((size_t)b * T_ + tr) * DD + d] = (__bf16)vv;
        sc[i] += avv[n] * FAST_TANH(qv[n] + vv);
      }
    }
    // Reduce over the 16 lanes holding the same row (sum of all 128 d's)
    #pragma unroll
    for (int i = 0; i < 8; ++i) {
      float s = sc[i];
      s += __shfl_xor(s, 8, 16);
      s += __shfl_xor(s, 4, 16);
      s += __shfl_xor(s, 2, 16);
      s += __shfl_xor(s, 1, 16);
      sc[i] = s;
    }
    if (l16 == 0) {
      #pragma unroll
      for (int i = 0; i < 8; ++i) {
        const int tr = tb + i + half * 8;
        score_out[b * T_ + tr] = sc[i] + sb;
      }
    }
  }
}

// ---------------------------------------------------------------------------
// K2: monotonic-attention scan + context. One block per batch row.
//  p = sigmoid(score); cp = exp(excl_cumsum(log(clip(1-p))));
//  align = p * cp * cumsum(prev / max(cp,1e-10)); context = align @ v
// ---------------------------------------------------------------------------
__global__ __launch_bounds__(256) void k_scan_ctx(
    const float* __restrict__ score, const float* __restrict__ prev_al,
    const __bf16* __restrict__ v_bf16, float* __restrict__ ctx_out,
    float* __restrict__ align_out) {
  __shared__ float s_scan[256];
  __shared__ float s_align[256];
  __shared__ float s_ctx[8][DD];  // 8 t-subset partials x 128 d

  const int b = blockIdx.x;
  const int tid = threadIdx.x;
  const int dg = (tid & 31) * 4;  // 4 contiguous d per thread
  const int hh = tid >> 5;        // t-subset (32 rows each) of the 256-chunk

  float ctx0 = 0.0f, ctx1 = 0.0f, ctx2 = 0.0f, ctx3 = 0.0f;
  float carry_log = 0.0f;
  float carry_r = 0.0f;

  for (int c0 = 0; c0 < T_; c0 += 256) {
    const int t = c0 + tid;
    const float s = score[b * T_ + t];
    const float p = FAST_RCP(1.0f + __expf(-s));
    float om = 1.0f - p;
    om = fminf(fmaxf(om, TINY_F), 1.0f);
    const float lg = __logf(om);

    // inclusive scan of lg over the chunk
    s_scan[tid] = lg;
    __syncthreads();
    for (int off = 1; off < 256; off <<= 1) {
      const float tv = (tid >= off) ? s_scan[tid - off] : 0.0f;
      __syncthreads();
      s_scan[tid] += tv;
      __syncthreads();
    }
    const float incl_lg = s_scan[tid];
    const float tot_lg  = s_scan[255];
    const float cp = __expf(carry_log + incl_lg - lg);  // exclusive cumprod(1-p)
    const float r  = prev_al[b * T_ + t] * FAST_RCP(fmaxf(cp, 1e-10f));
    __syncthreads();

    // inclusive scan of r over the chunk
    s_scan[tid] = r;
    __syncthreads();
    for (int off = 1; off < 256; off <<= 1) {
      const float tv = (tid >= off) ? s_scan[tid - off] : 0.0f;
      __syncthreads();
      s_scan[tid] += tv;
      __syncthreads();
    }
    const float incl_r = s_scan[tid];
    const float tot_r  = s_scan[255];

    const float align = p * cp * (carry_r + incl_r);
    s_align[tid] = align;
    align_out[b * T_ + t] = align;
    carry_log += tot_lg;
    carry_r   += tot_r;
    __syncthreads();

    // context accumulation: thread covers 4 contiguous d over 32 t-rows
    const __bf16* vb =
        v_bf16 + ((size_t)b * T_ + c0 + hh * 32) * DD + dg;
    #pragma unroll 4
    for (int j = 0; j < 32; ++j) {
      const v4bf x = *(const v4bf*)(vb + (size_t)j * DD);  // 8B aligned
      const float a = s_align[hh * 32 + j];
      ctx0 = fmaf(a, (float)x[0], ctx0);
      ctx1 = fmaf(a, (float)x[1], ctx1);
      ctx2 = fmaf(a, (float)x[2], ctx2);
      ctx3 = fmaf(a, (float)x[3], ctx3);
    }
    __syncthreads();
  }

  s_ctx[hh][dg + 0] = ctx0;
  s_ctx[hh][dg + 1] = ctx1;
  s_ctx[hh][dg + 2] = ctx2;
  s_ctx[hh][dg + 3] = ctx3;
  __syncthreads();
  if (tid < DD) {
    float c = 0.0f;
    #pragma unroll
    for (int h = 0; h < 8; ++h) c += s_ctx[h][tid];
    ctx_out[b * DD + tid] = c;
  }
}

// ---------------------------------------------------------------------------
extern "C" void kernel_launch(void* const* d_in, const int* in_sizes, int n_in,
                              void* d_out, int out_size, void* d_ws, size_t ws_size,
                              hipStream_t stream) {
  const float* query = (const float*)d_in[0];  // [B, Qd]
  const float* value = (const float*)d_in[1];  // [B, T, Vd]
  const float* prev  = (const float*)d_in[2];  // [B, T]
  const float* Wq    = (const float*)d_in[3];  // [Qd, D]
  const float* bq    = (const float*)d_in[4];  // [D]
  const float* Wv    = (const float*)d_in[5];  // [Vd, D]
  const float* bv    = (const float*)d_in[6];  // [D]
  const float* av    = (const float*)d_in[7];  // [D]
  const float* sb    = (const float*)d_in[8];  // scalar

  float* out = (float*)d_out;            // [B*D context][B*T alignment]
  char*  ws  = (char*)d_ws;

  float*  q_ws     = (float*)ws;                                  // 16 KB
  float*  score_ws = (float*)(ws + (size_t)B_ * DD * 4);          // 1 MB
  __bf16* v_ws     = (__bf16*)(ws + (size_t)B_ * DD * 4
                                  + (size_t)B_ * T_ * 4);         // 64 MB

  k_q<<<dim3(B_), dim3(DD), 0, stream>>>(query, Wq, bq, q_ws);
  k_v_score<<<dim3(T_ / 256, B_), dim3(256), 0, stream>>>(
      value, Wv, bv, av, q_ws, sb, v_ws, score_ws);
  k_scan_ctx<<<dim3(B_), dim3(256), 0, stream>>>(
      score_ws, prev, v_ws, out, out + B_ * DD);
}
